// Multi_Query_Attention_57423712748133
// MI455X (gfx1250) — matrix operand order repaired and checked by reading, not compiled, not run on hardware
//
#include <hip/hip_runtime.h>
#include <hip/hip_bf16.h>
#include <math.h>

// ---------------------------------------------------------------------------
// MQA block for MI455X (gfx1250, wave32, WMMA).
// All matmuls use v_wmma_f32_16x16x32_bf16 (bf16 inputs, f32 accumulate).
// V operand of the attention PV matmul is staged into LDS with the gfx1250
// async-LDS path (global_load_async_to_lds_b128 / ASYNCcnt, inline asm),
// overlapped with the whole S phase + softmax.
// ---------------------------------------------------------------------------

typedef __attribute__((ext_vector_type(16))) __bf16 v16bf;
typedef __attribute__((ext_vector_type(8)))  float  v8f;

static_assert(sizeof(v16bf) == 32, "v16bf must be 32 bytes");
static_assert(sizeof(v8f)  == 32, "v8f must be 32 bytes");

union Frag {
    v16bf v;
    uint4 q[2];
    unsigned short u[16];
};

#define EPSV 1e-5f

#if defined(__gfx1250__)
#define HAVE_ASYNC_LDS 1
#else
#define HAVE_ASYNC_LDS 0
#endif

// copy 16 bytes global -> LDS asynchronously (tracked by ASYNCcnt)
static __device__ __forceinline__ void async_copy_b128(const void* g, void* l) {
#if HAVE_ASYNC_LDS
    // low 32 bits of a generic LDS pointer are the LDS byte offset
    unsigned int loff = (unsigned int)(uintptr_t)l;
    asm volatile("global_load_async_to_lds_b128 %0, %1, off"
                 :
                 : "v"(loff), "v"(g)
                 : "memory");
#else
    *(uint4*)l = *(const uint4*)g;
#endif
}

static __device__ __forceinline__ void wait_async0() {
#if HAVE_ASYNC_LDS
    asm volatile("s_wait_asynccnt 0x0" ::: "memory");
#endif
}

// float -> bf16 (round to nearest even-ish)
static __device__ __forceinline__ unsigned short f2bf(float f) {
    union { float f; unsigned u; } c; c.f = f;
    unsigned r = c.u + 0x7FFFu + ((c.u >> 16) & 1u);
    return (unsigned short)(r >> 16);
}

static __device__ __forceinline__ v8f wmma_bf16(const Frag& a, const Frag& b, v8f c) {
    // (neg_a, A, neg_b, B, c_mod, C, reuse_a, reuse_b)
    return __builtin_amdgcn_wmma_f32_16x16x32_bf16(false, a.v, false, b.v,
                                                   (short)0, c, false, false);
}

static __device__ __forceinline__ float silu(float y) {
    return y * (1.0f / (1.0f + __expf(-y)));
}

// ---------------------------------------------------------------------------
// Generic GEMM + BN + SiLU:  Y[o,p] = silu(bn(sum_c W[o,c] * X[c,p])), per batch.
// X is the batch's buffer raw-reinterpreted as (Cin, 1024) row-major.
// MODE 0: scatter Q      -> out_a bf16  q[r][b][h][n][d]   (scale folded in)
// MODE 1: scatter K / V  -> out_a/out_b bf16  [b][h][n][d]
// MODE 2: final output   -> out_f f32   flat [b][o][p]
// grid = (Cout/128, 1024/128, B), block = 256 (8 waves)
// ---------------------------------------------------------------------------
template <int MODE>
__global__ __launch_bounds__(256)
void gemm_bn_silu_kernel(const float* __restrict__ X, const float* __restrict__ W,
                         const float* __restrict__ bn_g, const float* __restrict__ bn_b,
                         const float* __restrict__ bn_m, const float* __restrict__ bn_v,
                         float extra_scale, int Cin,
                         unsigned short* __restrict__ out_a,
                         unsigned short* __restrict__ out_b,
                         float* __restrict__ out_f) {
    __shared__ unsigned short sW[128 * 32];   // [m][k]  bf16, 8 KB
    __shared__ unsigned short sXt[128 * 32];  // [n][k]  bf16 (transposed), 8 KB

    const int tid  = threadIdx.x;
    const int lane = tid & 31;
    const int wave = tid >> 5;
    const int half = lane >> 4;
    const int l16  = lane & 15;

    const int bo0 = blockIdx.x * 128;  // output-channel base
    const int po0 = blockIdx.y * 128;  // pixel base
    const int bat = blockIdx.z;

    const float* Xb = X + (size_t)bat * (size_t)Cin * 1024;

    const int wm = wave >> 1;  // 0..3 : row block of 32
    const int wn = wave & 1;   // 0..1 : col block of 64

    v8f acc[2][4];
#pragma unroll
    for (int i = 0; i < 2; ++i)
#pragma unroll
        for (int j = 0; j < 4; ++j) acc[i][j] = (v8f){0,0,0,0,0,0,0,0};

    for (int kk = 0; kk < Cin; kk += 32) {
        // stage W tile (128 x 32), coalesced over k
#pragma unroll
        for (int i = 0; i < 16; ++i) {
            int idx = i * 256 + tid;
            int row = idx >> 5, k = idx & 31;
            sW[idx] = f2bf(W[(size_t)(bo0 + row) * Cin + (kk + k)]);
        }
        // stage X tile (32 x 128) transposed, coalesced over n
#pragma unroll
        for (int i = 0; i < 16; ++i) {
            int idx = i * 256 + tid;
            int k = idx >> 7, n = idx & 127;
            sXt[n * 32 + k] = f2bf(Xb[(size_t)(kk + k) * 1024 + po0 + n]);
        }
        // prefetch next K-slab (gfx1250 global_prefetch path)
        if (kk + 32 < Cin) {
            __builtin_prefetch(&W[(size_t)(bo0 + (tid & 127)) * Cin + kk + 32], 0, 0);
            __builtin_prefetch(&Xb[(size_t)(kk + 32 + (tid >> 3)) * 1024 + po0 + (tid & 7) * 16], 0, 0);
        }
        __syncthreads();

        Frag af[2], bf[4];
#pragma unroll
        for (int ti = 0; ti < 2; ++ti) {
            int mrow = wm * 32 + ti * 16 + l16;
            af[ti].q[0] = *(const uint4*)&sW[mrow * 32 + half * 8];
            af[ti].q[1] = *(const uint4*)&sW[mrow * 32 + 16 + half * 8];
        }
#pragma unroll
        for (int tj = 0; tj < 4; ++tj) {
            int ncol = wn * 64 + tj * 16 + l16;
            bf[tj].q[0] = *(const uint4*)&sXt[ncol * 32 + half * 8];
            bf[tj].q[1] = *(const uint4*)&sXt[ncol * 32 + 16 + half * 8];
        }
#pragma unroll
        for (int ti = 0; ti < 2; ++ti)
#pragma unroll
            for (int tj = 0; tj < 4; ++tj)
                acc[ti][tj] = wmma_bf16(af[ti], bf[tj], acc[ti][tj]);
        __syncthreads();
    }

    // epilogue: BN + SiLU + scatter
#pragma unroll
    for (int ti = 0; ti < 2; ++ti) {
#pragma unroll
        for (int j = 0; j < 8; ++j) {
            int o = bo0 + wm * 32 + ti * 16 + half * 8 + j;
            float sc = bn_g[o] * rsqrtf(bn_v[o] + EPSV);
            float bi = bn_b[o] - bn_m[o] * sc;
#pragma unroll
            for (int tj = 0; tj < 4; ++tj) {
                int p = po0 + wn * 64 + tj * 16 + l16;
                float y = acc[ti][tj][j] * sc + bi;
                y = silu(y) * extra_scale;
                if (MODE == 0) {
                    // q4[b,o,p] -> (r,b,h,n,dd)
                    int n   = o * 2 + (p >> 9);
                    int rem = p & 511;
                    int h   = rem >> 7;
                    int dd  = (rem >> 2) & 31;
                    int r   = rem & 3;
                    out_a[((((size_t)r * 4 + bat) * 4 + h) * 1024 + n) * 32 + dd] = f2bf(y);
                } else if (MODE == 1) {
                    // kv4[b,o,p] -> (s,b,h,n,dd)
                    int n   = o * 4 + (p >> 8);
                    int rem = p & 255;
                    int s   = rem >> 7;
                    int h   = (rem >> 5) & 3;
                    int dd  = rem & 31;
                    unsigned short* dst = s ? out_b : out_a;
                    dst[(((size_t)bat * 4 + h) * 1024 + n) * 32 + dd] = f2bf(y);
                } else {
                    out_f[(size_t)bat * 524288 + (size_t)o * 1024 + p] = y;
                }
            }
        }
    }
}

// ---------------------------------------------------------------------------
// Attention: one WG per (r,b,h, 32-query tile).
//   V (64KB bf16) async-staged to LDS at kernel start (overlapped with S).
//   S (32x1024) via WMMA -> LDS f32 ; softmax (max + exp, sum deferred) ;
//   P bf16 in LDS ; O = P*V via WMMA, cross-wave LDS reduce ; scatter with
//   the reference's concat/reshape index math into obuf (B,N,C) f32.
// grid = (32, 64), block = 256, dyn-LDS = 262272 B
// ---------------------------------------------------------------------------
__global__ __launch_bounds__(256)
void mqa_attn_kernel(const unsigned short* __restrict__ qbuf,
                     const unsigned short* __restrict__ kbuf,
                     const unsigned short* __restrict__ vbuf,
                     float* __restrict__ obuf) {
    extern __shared__ char smem[];
    float*          sS   = (float*)smem;                                  // 32*1024 f32   (128K)
    unsigned short* sP   = (unsigned short*)(smem + 131072);              // 32*1024 bf16  ( 64K)
    unsigned short* sV   = (unsigned short*)(smem + 131072 + 65536);      // 1024*32 bf16  ( 64K)
    float*          sRow = (float*)(smem + 131072 + 65536 + 65536);       // 32 f32
    float*          sPart = sS;  // reused for partial O (8*32*32 f32)

    const int tid  = threadIdx.x;
    const int lane = tid & 31;
    const int wave = tid >> 5;
    const int half = lane >> 4;
    const int l16  = lane & 15;

    const int n0 = blockIdx.x * 32;
    const int g  = blockIdx.y;          // r*16 + b*4 + h
    const int r  = g >> 4;
    const int b  = (g >> 2) & 3;
    const int h  = g & 3;

    const unsigned short* Q = qbuf + ((((size_t)r * 4 + b) * 4 + h) * 1024 + n0) * 32;
    const unsigned short* K = kbuf + (((size_t)b * 4 + h) * 1024) * 32;
    const unsigned short* V = vbuf + (((size_t)b * 4 + h) * 1024) * 32;

    // ---- stage V (1024x32 bf16, 64KB) into LDS; async path overlaps with S ----
#pragma unroll
    for (int i = 0; i < 16; ++i) {
        int e = tid * 8 + i * 2048;  // u16 elements, 16B per transfer per lane
        async_copy_b128(V + e, sV + e);
    }

    // Q A-fragments for rows n0..n0+31 (scale already folded in)
    Frag qa[2];
#pragma unroll
    for (int ti = 0; ti < 2; ++ti) {
        int mrow = ti * 16 + l16;
        qa[ti].q[0] = *(const uint4*)&Q[mrow * 32 + half * 8];
        qa[ti].q[1] = *(const uint4*)&Q[mrow * 32 + 16 + half * 8];
    }

    // ---- S phase: wave handles key columns [wave*128, wave*128+128) ----
    for (int ct = 0; ct < 8; ++ct) {
        int mcb = wave * 128 + ct * 16;
        Frag kb;
        kb.q[0] = *(const uint4*)&K[(mcb + l16) * 32 + half * 8];
        kb.q[1] = *(const uint4*)&K[(mcb + l16) * 32 + 16 + half * 8];
#pragma unroll
        for (int ti = 0; ti < 2; ++ti) {
            v8f c = (v8f){0,0,0,0,0,0,0,0};
            c = wmma_bf16(qa[ti], kb, c);
#pragma unroll
            for (int j = 0; j < 8; ++j) {
                int row = ti * 16 + half * 8 + j;
                sS[row * 1024 + mcb + l16] = c[j];
            }
        }
    }
    __syncthreads();

    // ---- softmax: wave handles rows 4*wave .. 4*wave+3 ----
    for (int rr = 0; rr < 4; ++rr) {
        int row = wave * 4 + rr;
        float mx = -1e30f;
#pragma unroll
        for (int i = 0; i < 32; ++i) mx = fmaxf(mx, sS[row * 1024 + lane + i * 32]);
#pragma unroll
        for (int m = 16; m >= 1; m >>= 1) mx = fmaxf(mx, __shfl_xor(mx, m, 32));
        float sum = 0.f;
#pragma unroll
        for (int i = 0; i < 32; ++i) {
            float e = __expf(sS[row * 1024 + lane + i * 32] - mx);
            sP[row * 1024 + lane + i * 32] = f2bf(e);
            sum += e;
        }
#pragma unroll
        for (int m = 16; m >= 1; m >>= 1) sum += __shfl_xor(sum, m, 32);
        if (lane == 0) sRow[row] = sum;
    }
    wait_async0();   // V staging complete (this wave's transfers)
    __syncthreads(); // all waves' transfers + sP visible

    // ---- O phase: wave accumulates key chunks {wave, wave+8, wave+16, wave+24} ----
    v8f oc[2][2];
#pragma unroll
    for (int i = 0; i < 2; ++i)
#pragma unroll
        for (int j = 0; j < 2; ++j) oc[i][j] = (v8f){0,0,0,0,0,0,0,0};

    for (int s = 0; s < 4; ++s) {
        int c0 = (wave + s * 8) * 32;
        Frag pa[2], vb[2];
#pragma unroll
        for (int ti = 0; ti < 2; ++ti) {
            int row = ti * 16 + l16;
            pa[ti].q[0] = *(const uint4*)&sP[row * 1024 + c0 + half * 8];
            pa[ti].q[1] = *(const uint4*)&sP[row * 1024 + c0 + 16 + half * 8];
        }
#pragma unroll
        for (int tj = 0; tj < 2; ++tj) {
            int dd = tj * 16 + l16;
#pragma unroll
            for (int e = 0; e < 16; ++e)
                vb[tj].u[e] = sV[(c0 + (e < 8 ? half * 8 + e : 16 + half * 8 + (e - 8))) * 32 + dd];
        }
#pragma unroll
        for (int ti = 0; ti < 2; ++ti)
#pragma unroll
            for (int tj = 0; tj < 2; ++tj)
                oc[ti][tj] = wmma_bf16(pa[ti], vb[tj], oc[ti][tj]);
    }

    // partial O -> LDS (sS region is free now)
#pragma unroll
    for (int ti = 0; ti < 2; ++ti)
#pragma unroll
        for (int tj = 0; tj < 2; ++tj)
#pragma unroll
            for (int j = 0; j < 8; ++j) {
                int row = ti * 16 + half * 8 + j;
                int dd  = tj * 16 + l16;
                sPart[wave * 1024 + row * 32 + dd] = oc[ti][tj][j];
            }
    __syncthreads();

    // ---- cross-wave reduce + 1/rowsum + scatter (concat/reshape mapping) ----
#pragma unroll
    for (int i = 0; i < 4; ++i) {
        int idx = tid + i * 256;           // 0..1023
        int row = idx >> 5, dd = idx & 31;
        float s = 0.f;
#pragma unroll
        for (int w = 0; w < 8; ++w) s += sPart[w * 1024 + idx];
        s /= sRow[row];
        int n  = n0 + row;
        int n2 = h * 256 + (n >> 2);
        int ch = r * 128 + ((n & 3) << 5) + dd;
        obuf[(size_t)b * 524288 + (size_t)n2 * 512 + ch] = s;
    }
}

// ---------------------------------------------------------------------------
// Host side
// ---------------------------------------------------------------------------
extern "C" void kernel_launch(void* const* d_in, const int* in_sizes, int n_in,
                              void* d_out, int out_size, void* d_ws, size_t ws_size,
                              hipStream_t stream) {
    (void)in_sizes; (void)n_in; (void)out_size; (void)ws_size;
    const float* x      = (const float*)d_in[0];
    const float* y      = (const float*)d_in[1];
    const float* q_w    = (const float*)d_in[2];
    const float* q_g    = (const float*)d_in[3];
    const float* q_b    = (const float*)d_in[4];
    const float* q_m    = (const float*)d_in[5];
    const float* q_v    = (const float*)d_in[6];
    const float* kv_w   = (const float*)d_in[7];
    const float* kv_g   = (const float*)d_in[8];
    const float* kv_b   = (const float*)d_in[9];
    const float* kv_m   = (const float*)d_in[10];
    const float* kv_v   = (const float*)d_in[11];
    const float* pw     = (const float*)d_in[12];
    const float* pg     = (const float*)d_in[13];
    const float* pb     = (const float*)d_in[14];
    const float* pm     = (const float*)d_in[15];
    const float* pv     = (const float*)d_in[16];
    float* out = (float*)d_out;

    char* ws = (char*)d_ws;
    unsigned short* qbuf = (unsigned short*)(ws);                 // 4 MB : (CR,B,NH,N,d) bf16
    unsigned short* kbuf = (unsigned short*)(ws + (4u << 20));    // 1 MB : (B,NH,N,d) bf16
    unsigned short* vbuf = (unsigned short*)(ws + (5u << 20));    // 1 MB
    float*          att  = (float*)(ws + (6u << 20));             // 8 MB : (B,N,C) f32

    const float scale = 0.08838834764831845f;  // (C/NH)^-0.5 = 128^-0.5

    // Q projection: Cout=512
    gemm_bn_silu_kernel<0><<<dim3(4, 8, 4), 256, 0, stream>>>(
        x, q_w, q_g, q_b, q_m, q_v, scale, 512, qbuf, nullptr, nullptr);
    // KV projection: Cout=256
    gemm_bn_silu_kernel<1><<<dim3(2, 8, 4), 256, 0, stream>>>(
        y, kv_w, kv_g, kv_b, kv_m, kv_v, 1.0f, 512, kbuf, vbuf, nullptr);
    // Attention (64 groups x 32 query tiles)
    const size_t smem = 131072 + 65536 + 65536 + 128;  // 262272 B (<320KB WGP budget)
    mqa_attn_kernel<<<dim3(32, 64), 256, smem, stream>>>(qbuf, kbuf, vbuf, att);
    // Output projection: Cout=512, result straight to d_out
    gemm_bn_silu_kernel<2><<<dim3(4, 8, 4), 256, 0, stream>>>(
        att, pw, pg, pb, pm, pv, 1.0f, 512, nullptr, nullptr, out);
}
